// UNet_config1_21646635171981
// MI455X (gfx1250) — matrix operand order repaired and checked by reading, not compiled, not run on hardware
//
#include <hip/hip_runtime.h>
#include <hip/hip_bf16.h>

typedef __attribute__((ext_vector_type(16))) _Float16 v16h;
typedef __attribute__((ext_vector_type(8)))  float    v8f;

// ---------------------------------------------------------------------------
// Dynamic per-sample conv: x[B,1,Hin,Win], w[B,Cout,8,8], pad=3 -> [B,Cout,255,255]
// Fused PReLU (scalar alpha). ~1e9 MACs -> plain VALU kernel, bandwidth bound.
// ---------------------------------------------------------------------------
__global__ void k_dyn_conv_prelu(const float* __restrict__ x, const float* __restrict__ w,
                                 const float* __restrict__ alpha, float* __restrict__ out,
                                 unsigned total, int Cout, int Hin, int Win, int Ho, int Wo) {
    unsigned idx = blockIdx.x * blockDim.x + threadIdx.x;
    if (idx >= total) return;
    unsigned ox = idx % Wo; unsigned t = idx / Wo;
    unsigned oy = t % Ho;   t /= Ho;
    unsigned co = t % Cout; unsigned b = t / Cout;
    const float* wp = w + (b * Cout + co) * 64u;
    const float* xp = x + b * (unsigned)(Hin * Win);
    float acc = 0.f;
    #pragma unroll
    for (int ky = 0; ky < 8; ++ky) {
        int iy = (int)oy + ky - 3;
        if ((unsigned)iy >= (unsigned)Hin) continue;
        #pragma unroll
        for (int kx = 0; kx < 8; ++kx) {
            int ix = (int)ox + kx - 3;
            if ((unsigned)ix >= (unsigned)Win) continue;
            acc += xp[iy * Win + ix] * wp[ky * 8 + kx];
        }
    }
    float a = alpha[0];
    out[idx] = acc >= 0.f ? acc : a * acc;
}

// ---------------------------------------------------------------------------
// Implicit-GEMM conv via WMMA f32_16x16x32_f16 (wave32), M- and N-tiled.
// Each wave: MT co-tiles (MT*16 out channels) x NT pixel-tiles (NT*16 pixels).
//  - B (activation) fragments are built once per K-chunk and reused against
//    MT weight fragments (activation gather traffic / MT).
//  - A (weight) fragments are reused across NT pixel tiles (weight traffic /NT),
//    loaded as aligned float4 (two contiguous 8-float runs per lane).
//  - reuse_a hint set on the second WMMA of each A-sharing pair (ISA 7.12).
//   A (16x32 MxK f16):  m = lane&15,  k(e) = (e>>3)*16 + (lane>>4)*8 + (e&7)
//   B (32x16 KxN f16):  n = lane&15,  k(e) = (e>>3)*16 + (lane>>4)*8 + (e&7)
//   C/D (16x16 f32):    m = r + 8*(lane>>4), n = lane&15
// Spatial decode is shift/mask (Ho, Wo powers of two); EXEC stays all-ones.
// ---------------------------------------------------------------------------
template <int KH, int KW, int PAD, int MT, int NT>
__global__ void k_conv_wmma(const float* __restrict__ in, const float* __restrict__ wgt,
                            const float* __restrict__ bias, const float* __restrict__ alpha,
                            float* __restrict__ out,
                            int Cin, int Hin, int Win, int Cout,
                            int WoShift, int WoMask, int HoMask, int HWShift) {
    const int lane = threadIdx.x & 31;
    const int wave = threadIdx.x >> 5;
    const int lhi  = lane >> 4;           // 0 or 1
    const int llo  = lane & 15;
    const int Ktot = Cin * KH * KW;

    const int co_base = blockIdx.y * (16 * MT);
    const unsigned nwave = (((blockIdx.x << 2) + wave) * NT) << 4;   // first pixel of wave

    // decode NT pixel tiles for this lane
    unsigned ox[NT], oy[NT], bb[NT], inBase[NT], spat[NT];
    #pragma unroll
    for (int j = 0; j < NT; ++j) {
        unsigned n = nwave + (unsigned)(j << 4) + (unsigned)llo;
        ox[j] = n & (unsigned)WoMask;
        oy[j] = (n >> WoShift) & (unsigned)HoMask;
        bb[j] = n >> HWShift;
        inBase[j] = bb[j] * (unsigned)Cin * (unsigned)(Hin * Win);
        spat[j] = (oy[j] << WoShift) + ox[j];
    }

    const float* __restrict__ wrow0 = wgt + (unsigned)(co_base + llo) * (unsigned)Ktot;
    const int kA0 = lhi * 8;              // A lane K offset within chunk
    const int kB0 = lhi * 16;             // unused: B gather uses kA0, same k order as A

    v8f acc[MT * NT];
    #pragma unroll
    for (int t = 0; t < MT * NT; ++t) acc[t] = (v8f){};

    for (int k0 = 0; k0 < Ktot; k0 += 32) {
        // prefetch next chunk of this lane's weight row (gfx1250 global_prefetch)
        if (k0 + 32 < Ktot) __builtin_prefetch(wrow0 + k0 + 32, 0, 3);

        // B: activation fragments (im2col on the fly, zero-padded borders).
        // k -> (ci,ky,kx) decode is shared across the NT pixel tiles.
        v16h bf[NT];
        #pragma unroll
        for (int j = 0; j < NT; ++j) bf[j] = (v16h){};
        #pragma unroll
        for (int e = 0; e < 16; ++e) {
            int k  = k0 + kA0 + (e & 7) + ((e >> 3) << 4);
            int ci = k / (KH * KW);
            int tt = k - ci * (KH * KW);
            int ky = tt / KW;
            int kx = tt - ky * KW;
            unsigned ciBase = (unsigned)(ci * Hin) * (unsigned)Win;
            #pragma unroll
            for (int j = 0; j < NT; ++j) {
                int iy = (int)oy[j] + ky - PAD;
                int ix = (int)ox[j] + kx - PAD;
                float v = 0.f;
                if ((unsigned)iy < (unsigned)Hin && (unsigned)ix < (unsigned)Win)
                    v = in[inBase[j] + ciBase + (unsigned)(iy * Win + ix)];
                bf[j][e] = (_Float16)v;
            }
        }

        // MT weight fragments (aligned float4 loads) x NT activation fragments
        #pragma unroll
        for (int t = 0; t < MT; ++t) {
            const float* __restrict__ wrow = wrow0 + (unsigned)(t * 16) * (unsigned)Ktot;
            const float4* __restrict__ q0p = (const float4*)(wrow + k0 + kA0);
            const float4* __restrict__ q1p = (const float4*)(wrow + k0 + 16 + kA0);
            float4 q0 = q0p[0], q1 = q0p[1];    // k = k0+kA0+0..7
            float4 q2 = q1p[0], q3 = q1p[1];    // k = k0+16+kA0+0..7
            v16h af;
            af[0]  = (_Float16)q0.x; af[1]  = (_Float16)q0.y;
            af[2]  = (_Float16)q0.z; af[3]  = (_Float16)q0.w;
            af[4]  = (_Float16)q1.x; af[5]  = (_Float16)q1.y;
            af[6]  = (_Float16)q1.z; af[7]  = (_Float16)q1.w;
            af[8]  = (_Float16)q2.x; af[9]  = (_Float16)q2.y;
            af[10] = (_Float16)q2.z; af[11] = (_Float16)q2.w;
            af[12] = (_Float16)q3.x; af[13] = (_Float16)q3.y;
            af[14] = (_Float16)q3.z; af[15] = (_Float16)q3.w;

            acc[t * NT + 0] = __builtin_amdgcn_wmma_f32_16x16x32_f16(
                false, af, false, bf[0], (short)0, acc[t * NT + 0],
                /*reuse_a=*/false, /*reuse_b=*/false);
            if (NT > 1)
                acc[t * NT + 1] = __builtin_amdgcn_wmma_f32_16x16x32_f16(
                    false, af, false, bf[1], (short)0, acc[t * NT + 1],
                    /*reuse_a=*/true, /*reuse_b=*/false);
        }
    }

    const float a = alpha[0];
    #pragma unroll
    for (int t = 0; t < MT; ++t) {
        #pragma unroll
        for (int r = 0; r < 8; ++r) {
            int co = co_base + t * 16 + r + (lhi << 3);
            float bs = bias[co];
            #pragma unroll
            for (int j = 0; j < NT; ++j) {
                float v = acc[t * NT + j][r] + bs;
                v = v >= 0.f ? v : a * v;
                out[((bb[j] * (unsigned)Cout + (unsigned)co) << HWShift) + spat[j]] = v;
            }
        }
    }
}

// ---------------------------------------------------------------------------
// BatchNorm (training mode, biased var, eps=1e-5): per-channel reduce + apply
// ---------------------------------------------------------------------------
__global__ void k_bn_reduce(const float* __restrict__ x, float* __restrict__ stats,
                            int B, int C, unsigned HW) {
    const int c = blockIdx.x;
    float s = 0.f, s2 = 0.f;
    for (int b = 0; b < B; ++b) {
        const float* p = x + (unsigned)(b * C + c) * HW;
        for (unsigned i = threadIdx.x; i < HW; i += blockDim.x) {
            float v = p[i];
            s += v; s2 += v * v;
        }
    }
    __shared__ float sh[256], sh2[256];
    sh[threadIdx.x] = s; sh2[threadIdx.x] = s2;
    __syncthreads();
    for (int ofs = blockDim.x >> 1; ofs > 0; ofs >>= 1) {
        if ((int)threadIdx.x < ofs) {
            sh[threadIdx.x]  += sh[threadIdx.x + ofs];
            sh2[threadIdx.x] += sh2[threadIdx.x + ofs];
        }
        __syncthreads();
    }
    if (threadIdx.x == 0) {
        float N = (float)B * (float)HW;
        float m = sh[0] / N;
        float var = sh2[0] / N - m * m;
        stats[2 * c]     = m;
        stats[2 * c + 1] = var;
    }
}

__global__ void k_bn_apply(float* __restrict__ x, const float* __restrict__ stats,
                           const float* __restrict__ g, const float* __restrict__ bt,
                           int HWShift, int CMask, unsigned total) {
    unsigned idx = blockIdx.x * blockDim.x + threadIdx.x;
    if (idx >= total) return;
    int c = (int)((idx >> HWShift) & (unsigned)CMask);
    float m = stats[2 * c], v = stats[2 * c + 1];
    x[idx] = (x[idx] - m) * rsqrtf(v + 1e-5f) * g[c] + bt[c];
}

// ---------------------------------------------------------------------------
// 2x2 max pool, stride 2 (H, W powers of two)
// ---------------------------------------------------------------------------
__global__ void k_maxpool2(const float* __restrict__ in, float* __restrict__ out,
                           unsigned total, int WoShift, int HoShift, int W) {
    unsigned idx = blockIdx.x * blockDim.x + threadIdx.x;
    if (idx >= total) return;
    unsigned ox = idx & ((1u << WoShift) - 1);
    unsigned oy = (idx >> WoShift) & ((1u << HoShift) - 1);
    unsigned bc = idx >> (WoShift + HoShift);
    const float* p = in + (bc << (WoShift + HoShift + 2)) + ((oy * 2) << (WoShift + 1)) + ox * 2;
    float v0 = p[0], v1 = p[1], v2 = p[W], v3 = p[W + 1];
    float m = v0 > v1 ? v0 : v1;
    float m2 = v2 > v3 ? v2 : v3;
    out[idx] = m > m2 ? m : m2;
}

// ---------------------------------------------------------------------------
// Bilinear x2 upsample (align_corners=True) into concat buffer [B,Ctot,2H,2W]
// at channel offset 0. H, W, C powers of two.
// ---------------------------------------------------------------------------
__global__ void k_up2_into(const float* __restrict__ in, float* __restrict__ out,
                           unsigned total, int H, int W, int WoShift, int HoShift,
                           int CShift, int Ctot) {
    unsigned idx = blockIdx.x * blockDim.x + threadIdx.x;
    if (idx >= total) return;
    unsigned xo = idx & ((1u << WoShift) - 1);
    unsigned yo = (idx >> WoShift) & ((1u << HoShift) - 1);
    unsigned c  = (idx >> (WoShift + HoShift)) & ((1u << CShift) - 1);
    unsigned b  = idx >> (WoShift + HoShift + CShift);
    int Ho = 2 * H, Wo = 2 * W;
    float fy = (float)yo * ((float)(H - 1) / (float)(Ho - 1));
    float fx = (float)xo * ((float)(W - 1) / (float)(Wo - 1));
    int y0 = (int)floorf(fy); int y1 = y0 + 1 < H ? y0 + 1 : H - 1; float wy = fy - (float)y0;
    int x0 = (int)floorf(fx); int x1 = x0 + 1 < W ? x0 + 1 : W - 1; float wx = fx - (float)x0;
    const float* p = in + (((b << CShift) + c) * (unsigned)(H * W));
    float top = p[y0 * W + x0] * (1.f - wx) + p[y0 * W + x1] * wx;
    float bot = p[y1 * W + x0] * (1.f - wx) + p[y1 * W + x1] * wx;
    out[((b * (unsigned)Ctot + c) << (WoShift + HoShift)) + (yo << WoShift) + xo] =
        top * (1.f - wy) + bot * wy;
}

// copy a skip tensor [B,C,H,W] into concat buffer [B,Ctot,H,W] at channel c0
__global__ void k_copy_into(const float* __restrict__ in, float* __restrict__ out,
                            unsigned total, int HWShift, int CShift, int Ctot, int c0) {
    unsigned idx = blockIdx.x * blockDim.x + threadIdx.x;
    if (idx >= total) return;
    unsigned p = idx & ((1u << HWShift) - 1);
    unsigned c = (idx >> HWShift) & ((1u << CShift) - 1);
    unsigned b = idx >> (HWShift + CShift);
    out[((b * (unsigned)Ctot + (unsigned)c0 + c) << HWShift) + p] = in[idx];
}

// final 1x1 conv, Cin -> 1  (HW = 65536)
__global__ void k_conv1x1_last(const float* __restrict__ in, const float* __restrict__ w,
                               const float* __restrict__ bias, float* __restrict__ out,
                               unsigned total, int Cin, int HWShift) {
    unsigned idx = blockIdx.x * blockDim.x + threadIdx.x;
    if (idx >= total) return;
    unsigned p = idx & ((1u << HWShift) - 1);
    unsigned b = idx >> HWShift;
    const float* ip = in + ((b * (unsigned)Cin) << HWShift) + p;
    float acc = bias[0];
    #pragma unroll 8
    for (int c = 0; c < Cin; ++c) acc += ip[(unsigned)c << HWShift] * w[c];
    out[idx] = acc;
}

// ---------------------------------------------------------------------------
static inline unsigned g1(unsigned n, int t) { return (n + t - 1) / t; }
static inline int ilog2(int v) { return 31 - __builtin_clz(v); }

extern "C" void kernel_launch(void* const* d_in, const int* in_sizes, int n_in,
                              void* d_out, int out_size, void* d_ws, size_t ws_size,
                              hipStream_t stream) {
    (void)in_sizes; (void)n_in; (void)out_size; (void)ws_size;
    const float* x     = (const float*)d_in[0];
    const float* dynw  = (const float*)d_in[1];
    const float* c1_w  = (const float*)d_in[2];  const float* c1_b  = (const float*)d_in[3];
    const float* d2_w1 = (const float*)d_in[4];  const float* d2_b1 = (const float*)d_in[5];
    const float* d2_w2 = (const float*)d_in[6];  const float* d2_b2 = (const float*)d_in[7];
    const float* d3_w1 = (const float*)d_in[8];  const float* d3_b1 = (const float*)d_in[9];
    const float* d3_w2 = (const float*)d_in[10]; const float* d3_b2 = (const float*)d_in[11];
    const float* d4_w1 = (const float*)d_in[12]; const float* d4_b1 = (const float*)d_in[13];
    const float* d4_w2 = (const float*)d_in[14]; const float* d4_b2 = (const float*)d_in[15];
    const float* u3_w1 = (const float*)d_in[16]; const float* u3_b1 = (const float*)d_in[17];
    const float* u3_w2 = (const float*)d_in[18]; const float* u3_b2 = (const float*)d_in[19];
    const float* u2_w1 = (const float*)d_in[20]; const float* u2_b1 = (const float*)d_in[21];
    const float* u2_w2 = (const float*)d_in[22]; const float* u2_b2 = (const float*)d_in[23];
    const float* u1_w1 = (const float*)d_in[24]; const float* u1_b1 = (const float*)d_in[25];
    const float* u1_w2 = (const float*)d_in[26]; const float* u1_b2 = (const float*)d_in[27];
    const float* lst_w = (const float*)d_in[28]; const float* lst_b = (const float*)d_in[29];
    const float* bn1_g = (const float*)d_in[30]; const float* bn1_b = (const float*)d_in[31];
    const float* bn2_g = (const float*)d_in[32]; const float* bn2_b = (const float*)d_in[33];
    const float* bn3_g = (const float*)d_in[34]; const float* bn3_b = (const float*)d_in[35];
    const float* bn4_g = (const float*)d_in[36]; const float* bn4_b = (const float*)d_in[37];
    const float* prelu = (const float*)d_in[38];
    float* yout = (float*)d_out;

    // workspace bump allocator (~720 MB total, buffers reused where lifetimes allow)
    char* ws = (char*)d_ws;
    size_t off = 0;
    auto alloc = [&](size_t bytes) -> float* {
        float* p = (float*)(ws + off);
        off += (bytes + 255) & ~(size_t)255;
        return p;
    };
    float* bufDyn = alloc(66585600ULL);   // [8,32,255,255]
    float* c1     = alloc(67108864ULL);   // [8,32,256,256]   (skip)
    float* c2     = alloc(67108864ULL);   // [8,128,128,128]  (skip)
    float* c3     = alloc(33554432ULL);   // [8,256,64,64]    (skip)
    float* pool   = alloc(16777216ULL);   // largest pool output
    float* mbuf   = alloc(67108864ULL);   // double-conv intermediate
    float* obuf   = alloc(67108864ULL);   // double-conv output (h4/u3o/u2o/u1o)
    float* cat    = alloc(335544320ULL);  // shared concat buffer (largest: [8,160,256,256])
    float* stats  = alloc(4096ULL);       // BN mean/var

    const int T = 256;
    const int B = 8;

    // 3x3 pad-1 conv + PReLU, square power-of-two H=W. 4 waves x 32 pixels/wave.
    auto conv3 = [&](const float* in, const float* wgt, const float* bias, int aidx,
                     float* out, int Cin, int H, int Cout) {
        int s = ilog2(H);
        unsigned Ntot = (unsigned)B << (2 * s);
        dim3 grid(Ntot / 128u, (unsigned)(Cout >= 64 ? Cout / 64 : Cout / 32));
        if (Cout >= 64)
            k_conv_wmma<3, 3, 1, 4, 2><<<grid, 128, 0, stream>>>(
                in, wgt, bias, prelu + aidx, out, Cin, H, H, Cout, s, H - 1, H - 1, 2 * s);
        else
            k_conv_wmma<3, 3, 1, 2, 2><<<grid, 128, 0, stream>>>(
                in, wgt, bias, prelu + aidx, out, Cin, H, H, Cout, s, H - 1, H - 1, 2 * s);
    };
    auto bn = [&](float* t, const float* g, const float* b, int C, int HWShift) {
        unsigned HW = 1u << HWShift;
        k_bn_reduce<<<dim3((unsigned)C), 256, 0, stream>>>(t, stats, B, C, HW);
        unsigned total = (unsigned)B * (unsigned)C * HW;
        k_bn_apply<<<dim3(g1(total, T)), T, 0, stream>>>(t, stats, g, b, HWShift, C - 1, total);
    };

    // ---- encoder ----
    // dynamic 8x8 conv (pad 3) + PReLU a0 : [8,1,256,256] -> [8,32,255,255]
    {
        unsigned total = 8u * 32u * 255u * 255u;
        k_dyn_conv_prelu<<<dim3(g1(total, T)), T, 0, stream>>>(
            x, dynw, prelu + 0, bufDyn, total, 32, 256, 256, 255, 255);
    }
    // 4x4 conv pad 2 + PReLU a1 : [8,32,255,255] -> [8,32,256,256]
    {
        unsigned Ntot = 8u << 16;
        dim3 grid(Ntot / 128u, 1);
        k_conv_wmma<4, 4, 2, 2, 2><<<grid, 128, 0, stream>>>(
            bufDyn, c1_w, c1_b, prelu + 1, c1, 32, 255, 255, 32, 8, 255, 255, 16);
    }
    bn(c1, bn1_g, bn1_b, 32, 16);

    k_maxpool2<<<dim3(g1(8u * 32 * 128 * 128, T)), T, 0, stream>>>(c1, pool, 8u * 32 * 128 * 128, 7, 7, 256);
    conv3(pool, d2_w1, d2_b1, 2, mbuf, 32, 128, 128);
    conv3(mbuf, d2_w2, d2_b2, 3, c2, 128, 128, 128);
    bn(c2, bn2_g, bn2_b, 128, 14);

    k_maxpool2<<<dim3(g1(8u * 128 * 64 * 64, T)), T, 0, stream>>>(c2, pool, 8u * 128 * 64 * 64, 6, 6, 128);
    conv3(pool, d3_w1, d3_b1, 4, mbuf, 128, 64, 256);
    conv3(mbuf, d3_w2, d3_b2, 5, c3, 256, 64, 256);
    bn(c3, bn3_g, bn3_b, 256, 12);

    k_maxpool2<<<dim3(g1(8u * 256 * 32 * 32, T)), T, 0, stream>>>(c3, pool, 8u * 256 * 32 * 32, 5, 5, 64);
    conv3(pool, d4_w1, d4_b1, 6, mbuf, 256, 32, 512);
    conv3(mbuf, d4_w2, d4_b2, 7, obuf, 512, 32, 512);
    bn(obuf, bn4_g, bn4_b, 512, 10);

    // ---- decoder ----
    // up(h4) ++ c3 -> [8,768,64,64]
    k_up2_into<<<dim3(g1(8u * 512 * 64 * 64, T)), T, 0, stream>>>(
        obuf, cat, 8u * 512 * 64 * 64, 32, 32, 6, 6, 9, 768);
    k_copy_into<<<dim3(g1(8u * 256 * 4096, T)), T, 0, stream>>>(
        c3, cat, 8u * 256 * 4096, 12, 8, 768, 512);
    conv3(cat, u3_w1, u3_b1, 8, mbuf, 768, 64, 256);
    conv3(mbuf, u3_w2, u3_b2, 9, obuf, 256, 64, 256);

    // up(u3o) ++ c2 -> [8,384,128,128]
    k_up2_into<<<dim3(g1(8u * 256 * 128 * 128, T)), T, 0, stream>>>(
        obuf, cat, 8u * 256 * 128 * 128, 64, 64, 7, 7, 8, 384);
    k_copy_into<<<dim3(g1(8u * 128 * 16384, T)), T, 0, stream>>>(
        c2, cat, 8u * 128 * 16384, 14, 7, 384, 256);
    conv3(cat, u2_w1, u2_b1, 10, mbuf, 384, 128, 128);
    conv3(mbuf, u2_w2, u2_b2, 11, obuf, 128, 128, 128);

    // up(u2o) ++ c1 -> [8,160,256,256]
    k_up2_into<<<dim3(g1(8u * 128 * 256 * 256, T)), T, 0, stream>>>(
        obuf, cat, 8u * 128 * 256 * 256, 128, 128, 8, 8, 7, 160);
    k_copy_into<<<dim3(g1(8u * 32 * 65536, T)), T, 0, stream>>>(
        c1, cat, 8u * 32 * 65536, 16, 5, 160, 128);
    conv3(cat, u1_w1, u1_b1, 12, mbuf, 160, 256, 32);
    conv3(mbuf, u1_w2, u1_b2, 13, obuf, 32, 256, 32);

    // final 1x1 conv -> d_out [8,1,256,256]
    k_conv1x1_last<<<dim3(g1(8u * 65536, T)), T, 0, stream>>>(
        obuf, lst_w, lst_b, yout, 8u * 65536, 32, 16);
}
